// PointNetClassification_78993038508271
// MI455X (gfx1250) — matrix operand order repaired and checked by reading, not compiled, not run on hardware
//
#include <hip/hip_runtime.h>
#include <math.h>

// ---------------------------------------------------------------------------
// PointNet++ classification pipeline for MI455X (gfx1250), wave32 + WMMA bf16.
//
// Pipeline (all on `stream`):
//   convert all MLP weights f32 -> zero-padded bf16 [Kpad,Npad] (ws)
//   fps<4096>  -> pos1[8,2048,3]
//   sa<4096,3> -> x1[8,2048,128]    (fused kNN64 + gather + MLP + masked max)
//   fps<2048>  -> pos2[8,512,3]
//   sa<2048,128> -> x2[8,512,256]
//   init gmax = -inf; sa3 blocks of 16 pts -> atomicMax into gmax[8,1024]
//   head: [8(pad16),1024] @ W chain -> logits -> log_softmax -> d_out[8,10]
// ---------------------------------------------------------------------------

typedef __attribute__((ext_vector_type(16))) __bf16 v16bf;
typedef __attribute__((ext_vector_type(8)))  float  v8f;

#define NTHREADS 256
#define NWAVES   8

// ---- WMMA helper: one 16x16 D-tile accumulation step over K=32 --------------
// A: LDS, row-major [Mrows, lda] bf16.  W: global, row-major [Kpad, ldw] bf16.
// A-fragment layout (ISA 7.12.2, 16-bit A 16x32): lane l: M = l&15,
//   K in {0..7,16..23} (l<16) or {8..15,24..31} (l>=16), pairs packed per VGPR.
// B-fragment layout: lane = N column (l&15), vector elem j = K row, same K order as A.
__device__ __forceinline__
v8f wmma_step(const __bf16* __restrict__ A, int lda, int mt, int ks,
              const __bf16* __restrict__ W, int ldw, int nt,
              v8f acc, int lane) {
  const int hh = lane >> 4, l15 = lane & 15;
  const __bf16* arow = A + (mt * 16 + l15) * lda + ks * 32 + 8 * hh;
  v16bf a;
#pragma unroll
  for (int j = 0; j < 16; ++j) {
    const int kk = ((j >= 8) ? 16 : 0) + (((j >> 1) & 3) << 1) + (j & 1);
    a[j] = arow[kk];
  }
  v16bf b;
#pragma unroll
  for (int j = 0; j < 16; ++j) {
    const int kk = ((j >= 8) ? 16 : 0) + (((j >> 1) & 3) << 1) + (j & 1);
    b[j] = W[(ks * 32 + 8 * hh + kk) * ldw + nt * 16 + l15];
  }
  return __builtin_amdgcn_wmma_f32_16x16x32_bf16(false, a, false, b,
                                                 (short)0, acc, false, false);
}

// ---- generic MLP layer: Hout = act(A @ W + bias), all tiles across 8 waves --
template <int MROWS, int KDIM, int NOUT, bool RELU>
__device__ void mlp_layer(const __bf16* __restrict__ A,   // LDS [MROWS,KDIM]
                          const __bf16* __restrict__ W,   // global [KDIM,NOUT]
                          const float*  __restrict__ bias,
                          __bf16* __restrict__ Hout) {    // LDS [MROWS,NOUT]
  static_assert(MROWS % 16 == 0 && KDIM % 32 == 0 && NOUT % 16 == 0, "");
  constexpr int MT = MROWS / 16, NT = NOUT / 16, KT = KDIM / 32;
  const int wv = threadIdx.x >> 5, lane = threadIdx.x & 31;
  const int hh = lane >> 4, l15 = lane & 15;
  for (int t = wv; t < MT * NT; t += NWAVES) {
    const int mt = t / NT, nt = t % NT;
    v8f acc = {0.f, 0.f, 0.f, 0.f, 0.f, 0.f, 0.f, 0.f};
    for (int ks = 0; ks < KT; ++ks)
      acc = wmma_step(A, KDIM, mt, ks, W, NOUT, nt, acc, lane);
    const float bn = bias[nt * 16 + l15];
#pragma unroll
    for (int i = 0; i < 8; ++i) {
      float v = acc[i] + bn;
      if (RELU) v = fmaxf(v, 0.f);
      Hout[(mt * 16 + i + 8 * hh) * NOUT + nt * 16 + l15] = (__bf16)v;
    }
  }
}

// ---- SA final layer fused with radius mask + max over 64 neighbor rows ------
template <int KDIM, int NOUT>
__device__ void sa_final_layer(const __bf16* __restrict__ A,  // LDS [64,KDIM]
                               const __bf16* __restrict__ W,
                               const float*  __restrict__ bias,
                               const float*  __restrict__ nbrd2, float r2,
                               float* __restrict__ xout) {    // global [NOUT]
  constexpr int MT = 4, NT = NOUT / 16, KT = KDIM / 32;
  const int wv = threadIdx.x >> 5, lane = threadIdx.x & 31;
  const int hh = lane >> 4, l15 = lane & 15;
  for (int nt = wv; nt < NT; nt += NWAVES) {
    const float bn = bias[nt * 16 + l15];
    float mx = -INFINITY;
    for (int mt = 0; mt < MT; ++mt) {
      v8f acc = {0.f, 0.f, 0.f, 0.f, 0.f, 0.f, 0.f, 0.f};
      for (int ks = 0; ks < KT; ++ks)
        acc = wmma_step(A, KDIM, mt, ks, W, NOUT, nt, acc, lane);
#pragma unroll
      for (int i = 0; i < 8; ++i) {
        const int row = mt * 16 + i + 8 * hh;      // neighbor index 0..63
        if (nbrd2[row] <= r2) mx = fmaxf(mx, acc[i] + bn);
      }
    }
    mx = fmaxf(mx, __shfl_xor(mx, 16, 32));        // combine lane halves
    if (lane < 16)                                  // empty ball -> 0 (PyG)
      xout[nt * 16 + lane] = (mx == -INFINITY) ? 0.f : mx;
  }
}

__device__ __forceinline__ void atomicMaxFloat(float* addr, float val) {
  unsigned* ua = (unsigned*)addr;
  unsigned old = *ua;
  for (;;) {
    const float f = __uint_as_float(old);
    if (f >= val) break;
    const unsigned assumed = old;
    old = atomicCAS(ua, assumed, __float_as_uint(val));
    if (old == assumed) break;
  }
}

// ---- Global-SA final layer: per-column max over 16 rows -> atomicMax --------
template <int KDIM, int NOUT>
__device__ void global_final_layer(const __bf16* __restrict__ A, // LDS [16,KDIM]
                                   const __bf16* __restrict__ W,
                                   const float*  __restrict__ bias,
                                   float* __restrict__ gmax) {   // global [NOUT]
  constexpr int NT = NOUT / 16, KT = KDIM / 32;
  const int wv = threadIdx.x >> 5, lane = threadIdx.x & 31, l15 = lane & 15;
  for (int nt = wv; nt < NT; nt += NWAVES) {
    v8f acc = {0.f, 0.f, 0.f, 0.f, 0.f, 0.f, 0.f, 0.f};
    for (int ks = 0; ks < KT; ++ks)
      acc = wmma_step(A, KDIM, 0, ks, W, NOUT, nt, acc, lane);
    const float bn = bias[nt * 16 + l15];
    float mx = -INFINITY;
#pragma unroll
    for (int i = 0; i < 8; ++i) mx = fmaxf(mx, acc[i] + bn);   // 8 rows
    mx = fmaxf(mx, __shfl_xor(mx, 16, 32));                    // other 8 rows
    if (lane < 16) atomicMaxFloat(&gmax[nt * 16 + lane], mx);
  }
}

// ---------------------------------------------------------------------------
// Farthest point sampling: one block per cloud, serial over M picks.
// Tie-break = lowest index (matches jnp.argmax first-occurrence).
// ---------------------------------------------------------------------------
template <int NPTS>
__global__ __launch_bounds__(NTHREADS)
void fps_kernel(const float* __restrict__ pos, int M,
                float* __restrict__ posq) {
  const int b = blockIdx.x;
  __shared__ float mind[NPTS];
  __shared__ float sval[NTHREADS];
  __shared__ int   sidx[NTHREADS];
  __shared__ float curp[3];
  const float* P = pos + (size_t)b * NPTS * 3;
  const float x0 = P[0], y0 = P[1], z0 = P[2];
  for (int i = threadIdx.x; i < NPTS; i += NTHREADS) {
    const float dx = P[i * 3] - x0, dy = P[i * 3 + 1] - y0, dz = P[i * 3 + 2] - z0;
    mind[i] = dx * dx + dy * dy + dz * dz;
  }
  if (threadIdx.x == 0) {
    posq[(size_t)b * M * 3 + 0] = x0;
    posq[(size_t)b * M * 3 + 1] = y0;
    posq[(size_t)b * M * 3 + 2] = z0;
  }
  __syncthreads();
  for (int j = 1; j < M; ++j) {
    float best = -1.f; int bi = 0;
    for (int i = threadIdx.x; i < NPTS; i += NTHREADS) {
      const float d = mind[i];
      if (d > best) { best = d; bi = i; }
    }
    sval[threadIdx.x] = best; sidx[threadIdx.x] = bi;
    __syncthreads();
    for (int s = NTHREADS / 2; s > 0; s >>= 1) {
      if (threadIdx.x < s) {
        const float ov = sval[threadIdx.x + s]; const int oi = sidx[threadIdx.x + s];
        if (ov > sval[threadIdx.x] ||
            (ov == sval[threadIdx.x] && oi < sidx[threadIdx.x])) {
          sval[threadIdx.x] = ov; sidx[threadIdx.x] = oi;
        }
      }
      __syncthreads();
    }
    if (threadIdx.x == 0) {
      const int idx = sidx[0];
      curp[0] = P[idx * 3]; curp[1] = P[idx * 3 + 1]; curp[2] = P[idx * 3 + 2];
      posq[((size_t)b * M + j) * 3 + 0] = curp[0];
      posq[((size_t)b * M + j) * 3 + 1] = curp[1];
      posq[((size_t)b * M + j) * 3 + 2] = curp[2];
    }
    __syncthreads();
    const float cx = curp[0], cy = curp[1], cz = curp[2];
    for (int i = threadIdx.x; i < NPTS; i += NTHREADS) {
      const float dx = P[i * 3] - cx, dy = P[i * 3 + 1] - cy, dz = P[i * 3 + 2] - cz;
      mind[i] = fminf(mind[i], dx * dx + dy * dy + dz * dz);
    }
    __syncthreads();
  }
}

// ---------------------------------------------------------------------------
// Fused SA stage: one block per query point.
//   distances -> iterative 64-NN selection -> gather [x_j, pos_j - pos_i]
//   -> bf16 MLP (WMMA) -> radius-masked max over neighbors -> xout[C3].
// ---------------------------------------------------------------------------
template <int NPTS, int DINX, int DINPAD, int C1, int C2, int C3>
__global__ __launch_bounds__(NTHREADS)
void sa_kernel(const float* __restrict__ x,    // [B,NPTS,DINX]
               const float* __restrict__ pos,  // [B,NPTS,3]
               const float* __restrict__ posq, // [B,M,3]
               const __bf16* __restrict__ w1, const float* __restrict__ b1,
               const __bf16* __restrict__ w2, const float* __restrict__ b2,
               const __bf16* __restrict__ w3, const float* __restrict__ b3,
               float r2, int M,
               float* __restrict__ xout) {     // [B,M,C3]
  constexpr int KNB  = 64;
  constexpr int AMAX = (DINPAD > C2 ? DINPAD : C2);
  const int q = blockIdx.x % M, b = blockIdx.x / M;

  __shared__ float  dist[NPTS];
  __shared__ float  sval[NTHREADS];
  __shared__ int    sidx[NTHREADS];
  __shared__ int    nbr[KNB];
  __shared__ float  nbrd2[KNB];
  __shared__ __bf16 bufA[KNB * AMAX];   // holds A (layer1 in), then h2
  __shared__ __bf16 bufH[KNB * C1];     // holds h1

  __builtin_prefetch(w1, 0, 3);

  const float qx = posq[((size_t)b * M + q) * 3 + 0];
  const float qy = posq[((size_t)b * M + q) * 3 + 1];
  const float qz = posq[((size_t)b * M + q) * 3 + 2];

  // --- all pairwise squared distances to this query ---
  for (int i = threadIdx.x; i < NPTS; i += NTHREADS) {
    const float dx = pos[((size_t)b * NPTS + i) * 3 + 0] - qx;
    const float dy = pos[((size_t)b * NPTS + i) * 3 + 1] - qy;
    const float dz = pos[((size_t)b * NPTS + i) * 3 + 2] - qz;
    dist[i] = dx * dx + dy * dy + dz * dz;
  }
  __syncthreads();

  // --- iterative selection of the 64 nearest (stable, index tie-break) ---
  for (int k = 0; k < KNB; ++k) {
    float best = INFINITY; int bi = NPTS;
    for (int i = threadIdx.x; i < NPTS; i += NTHREADS) {
      const float d = dist[i];
      if (d < best) { best = d; bi = i; }
    }
    sval[threadIdx.x] = best; sidx[threadIdx.x] = bi;
    __syncthreads();
    for (int s = NTHREADS / 2; s > 0; s >>= 1) {
      if (threadIdx.x < s) {
        const float ov = sval[threadIdx.x + s]; const int oi = sidx[threadIdx.x + s];
        if (ov < sval[threadIdx.x] ||
            (ov == sval[threadIdx.x] && oi < sidx[threadIdx.x])) {
          sval[threadIdx.x] = ov; sidx[threadIdx.x] = oi;
        }
      }
      __syncthreads();
    }
    if (threadIdx.x == 0) {
      nbr[k] = sidx[0]; nbrd2[k] = sval[0];
      dist[sidx[0]] = INFINITY;
    }
    __syncthreads();
  }

  // --- build A = [x_j, pos_j - pos_i] as bf16, zero-padded to DINPAD ---
  for (int i = threadIdx.x; i < KNB * DINPAD; i += NTHREADS) {
    const int r = i / DINPAD, c = i % DINPAD;
    const int p = nbr[r];
    float v = 0.f;
    if (c < DINX) {
      v = x[((size_t)b * NPTS + p) * DINX + c];
    } else if (c < DINX + 3) {
      const int d = c - DINX;
      const float qc = (d == 0) ? qx : ((d == 1) ? qy : qz);
      v = pos[((size_t)b * NPTS + p) * 3 + d] - qc;
    }
    bufA[i] = (__bf16)v;
  }
  __syncthreads();

  mlp_layer<KNB, DINPAD, C1, true>(bufA, w1, b1, bufH);   // h1
  __syncthreads();
  mlp_layer<KNB, C1, C2, true>(bufH, w2, b2, bufA);       // h2 (overwrites A)
  __syncthreads();
  sa_final_layer<C2, C3>(bufA, w3, b3, nbrd2, r2,
                         xout + ((size_t)b * M + q) * C3);
}

// ---------------------------------------------------------------------------
// Global SA: blocks of 16 points; MLP [259->256->512->1024]; atomicMax pool.
// ---------------------------------------------------------------------------
__global__ __launch_bounds__(NTHREADS)
void sa3_kernel(const float* __restrict__ x2,   // [8,512,256]
                const float* __restrict__ pos2, // [8,512,3]
                const __bf16* __restrict__ w1, const float* __restrict__ b1,
                const __bf16* __restrict__ w2, const float* __restrict__ b2,
                const __bf16* __restrict__ w3, const float* __restrict__ b3,
                float* __restrict__ gmax) {     // [8,1024], pre-set to -inf
  constexpr int M2 = 512, DIN = 259, DINPAD = 288;
  const int blocksPerCloud = M2 / 16;
  const int b = blockIdx.x / blocksPerCloud;
  const int p0 = (blockIdx.x % blocksPerCloud) * 16;

  __shared__ __bf16 bufA[16 * DINPAD]; // 9 KB
  __shared__ __bf16 bufH1[16 * 256];   // 8 KB
  __shared__ __bf16 bufH2[16 * 512];   // 16 KB

  for (int i = threadIdx.x; i < 16 * DINPAD; i += NTHREADS) {
    const int r = i / DINPAD, c = i % DINPAD;
    const int p = p0 + r;
    float v = 0.f;
    if (c < 256)      v = x2[((size_t)b * M2 + p) * 256 + c];
    else if (c < DIN) v = pos2[((size_t)b * M2 + p) * 3 + (c - 256)];
    bufA[i] = (__bf16)v;
  }
  __syncthreads();
  mlp_layer<16, DINPAD, 256, true>(bufA, w1, b1, bufH1);
  __syncthreads();
  mlp_layer<16, 256, 512, true>(bufH1, w2, b2, bufH2);
  __syncthreads();
  global_final_layer<512, 1024>(bufH2, w3, b3, gmax + (size_t)b * 1024);
}

// ---------------------------------------------------------------------------
// Head: [8 (pad 16), 1024] -> 512 -> 256 -> 10 -> log_softmax.  One block.
// ---------------------------------------------------------------------------
__global__ __launch_bounds__(NTHREADS)
void head_kernel(const float* __restrict__ g,   // [8,1024]
                 const __bf16* __restrict__ w1, const float* __restrict__ b1,
                 const __bf16* __restrict__ w2, const float* __restrict__ b2,
                 const __bf16* __restrict__ w3, const float* __restrict__ b3,
                 float* __restrict__ out) {     // [8,10]
  __shared__ __bf16 bufA[16 * 1024];  // 32 KB
  __shared__ __bf16 bufH1[16 * 512];  // 16 KB
  __shared__ __bf16 bufH2[16 * 256];  // 8 KB
  __shared__ float  logits[16 * 16];

  for (int i = threadIdx.x; i < 16 * 1024; i += NTHREADS) {
    const int r = i / 1024, c = i % 1024;
    bufA[i] = (__bf16)((r < 8) ? g[r * 1024 + c] : 0.f);
  }
  __syncthreads();
  mlp_layer<16, 1024, 512, true>(bufA, w1, b1, bufH1);
  __syncthreads();
  mlp_layer<16, 512, 256, true>(bufH1, w2, b2, bufH2);
  __syncthreads();

  // final 16x16 tile (10 real cols), wave 0 only
  const int wv = threadIdx.x >> 5, lane = threadIdx.x & 31;
  const int hh = lane >> 4, l15 = lane & 15;
  if (wv == 0) {
    v8f acc = {0.f, 0.f, 0.f, 0.f, 0.f, 0.f, 0.f, 0.f};
    for (int ks = 0; ks < 256 / 32; ++ks)
      acc = wmma_step(bufH2, 256, 0, ks, w3, 16, 0, acc, lane);
    const float bn = (l15 < 10) ? b3[l15] : 0.f;
#pragma unroll
    for (int i = 0; i < 8; ++i)
      logits[(i + 8 * hh) * 16 + l15] = acc[i] + bn;
  }
  __syncthreads();

  if (threadIdx.x < 8) {
    const int r = threadIdx.x;
    float mx = -INFINITY;
    for (int c = 0; c < 10; ++c) mx = fmaxf(mx, logits[r * 16 + c]);
    float s = 0.f;
    for (int c = 0; c < 10; ++c) s += expf(logits[r * 16 + c] - mx);
    const float lse = mx + logf(s);
    for (int c = 0; c < 10; ++c) out[r * 10 + c] = logits[r * 16 + c] - lse;
  }
}

// ---------------------------------------------------------------------------
// Small utility kernels
// ---------------------------------------------------------------------------
__global__ void convert_w_kernel(const float* __restrict__ w, int din, int dout,
                                 __bf16* __restrict__ o, int dinp, int doutp) {
  const int i = blockIdx.x * blockDim.x + threadIdx.x;
  if (i >= dinp * doutp) return;
  const int r = i / doutp, c = i % doutp;
  o[i] = (__bf16)((r < din && c < dout) ? w[r * dout + c] : 0.f);
}

__global__ void fill_neginf_kernel(float* __restrict__ p, int n) {
  const int i = blockIdx.x * blockDim.x + threadIdx.x;
  if (i < n) p[i] = -INFINITY;
}

// ---------------------------------------------------------------------------
extern "C" void kernel_launch(void* const* d_in, const int* in_sizes, int n_in,
                              void* d_out, int out_size, void* d_ws, size_t ws_size,
                              hipStream_t stream) {
  (void)in_sizes; (void)n_in; (void)out_size; (void)ws_size;

  const float* x    = (const float*)d_in[0];   // [8,4096,3]
  const float* pos  = (const float*)d_in[1];   // [8,4096,3]
  // sa1 / sa2 / sa3 / head params, each (w,b) x 3 layers, flat in dict order
  const float* s1w[3] = {(const float*)d_in[2],  (const float*)d_in[4],  (const float*)d_in[6]};
  const float* s1b[3] = {(const float*)d_in[3],  (const float*)d_in[5],  (const float*)d_in[7]};
  const float* s2w[3] = {(const float*)d_in[8],  (const float*)d_in[10], (const float*)d_in[12]};
  const float* s2b[3] = {(const float*)d_in[9],  (const float*)d_in[11], (const float*)d_in[13]};
  const float* s3w[3] = {(const float*)d_in[14], (const float*)d_in[16], (const float*)d_in[18]};
  const float* s3b[3] = {(const float*)d_in[15], (const float*)d_in[17], (const float*)d_in[19]};
  const float* hw[3]  = {(const float*)d_in[20], (const float*)d_in[22], (const float*)d_in[24]};
  const float* hb[3]  = {(const float*)d_in[21], (const float*)d_in[23], (const float*)d_in[25]};

  // ---- workspace carve-out (256B aligned) ----
  char* ws = (char*)d_ws;
  size_t off = 0;
  auto carve = [&](size_t bytes) -> char* {
    char* p = ws + off;
    off = (off + bytes + 255) & ~(size_t)255;
    return p;
  };
  struct WSpec { const float* w; int din, dout, dinp, doutp; __bf16* dst; };
  WSpec wsp[12] = {
    {s1w[0],    6,   64,   32,   64, nullptr}, {s1w[1],   64,   64,   64,   64, nullptr},
    {s1w[2],   64,  128,   64,  128, nullptr},
    {s2w[0],  131,  128,  160,  128, nullptr}, {s2w[1],  128,  128,  128,  128, nullptr},
    {s2w[2],  128,  256,  128,  256, nullptr},
    {s3w[0],  259,  256,  288,  256, nullptr}, {s3w[1],  256,  512,  256,  512, nullptr},
    {s3w[2],  512, 1024,  512, 1024, nullptr},
    {hw[0],  1024,  512, 1024,  512, nullptr}, {hw[1],   512,  256,  512,  256, nullptr},
    {hw[2],   256,   10,  256,   16, nullptr},
  };
  for (int i = 0; i < 12; ++i)
    wsp[i].dst = (__bf16*)carve((size_t)wsp[i].dinp * wsp[i].doutp * sizeof(__bf16));

  float* pos1 = (float*)carve((size_t)8 * 2048 * 3   * sizeof(float));
  float* x1   = (float*)carve((size_t)8 * 2048 * 128 * sizeof(float));
  float* pos2 = (float*)carve((size_t)8 * 512 * 3    * sizeof(float));
  float* x2   = (float*)carve((size_t)8 * 512 * 256  * sizeof(float));
  float* gmax = (float*)carve((size_t)8 * 1024       * sizeof(float));

  // ---- weight conversion (tiny) ----
  for (int i = 0; i < 12; ++i) {
    const int tot = wsp[i].dinp * wsp[i].doutp;
    convert_w_kernel<<<(tot + 255) / 256, 256, 0, stream>>>(
        wsp[i].w, wsp[i].din, wsp[i].dout, wsp[i].dst, wsp[i].dinp, wsp[i].doutp);
  }

  // ---- SA1: FPS 4096->2048, fused conv ----
  fps_kernel<4096><<<8, NTHREADS, 0, stream>>>(pos, 2048, pos1);
  sa_kernel<4096, 3, 32, 64, 64, 128><<<8 * 2048, NTHREADS, 0, stream>>>(
      x, pos, pos1,
      wsp[0].dst, s1b[0], wsp[1].dst, s1b[1], wsp[2].dst, s1b[2],
      0.2f * 0.2f, 2048, x1);

  // ---- SA2: FPS 2048->512, fused conv ----
  fps_kernel<2048><<<8, NTHREADS, 0, stream>>>(pos1, 512, pos2);
  sa_kernel<2048, 128, 160, 128, 128, 256><<<8 * 512, NTHREADS, 0, stream>>>(
      x1, pos1, pos2,
      wsp[3].dst, s2b[0], wsp[4].dst, s2b[1], wsp[5].dst, s2b[2],
      0.4f * 0.4f, 512, x2);

  // ---- Global SA ----
  fill_neginf_kernel<<<(8 * 1024 + 255) / 256, 256, 0, stream>>>(gmax, 8 * 1024);
  sa3_kernel<<<8 * (512 / 16), NTHREADS, 0, stream>>>(
      x2, pos2,
      wsp[6].dst, s3b[0], wsp[7].dst, s3b[1], wsp[8].dst, s3b[2], gmax);

  // ---- Head + log_softmax ----
  head_kernel<<<1, NTHREADS, 0, stream>>>(
      gmax,
      wsp[9].dst, hb[0], wsp[10].dst, hb[1], wsp[11].dst, hb[2],
      (float*)d_out);
}
